// net_62139586838563
// MI455X (gfx1250) — matrix operand order repaired and checked by reading, not compiled, not run on hardware
//
#include <hip/hip_runtime.h>

typedef __attribute__((ext_vector_type(16))) _Float16 v16h;
typedef __attribute__((ext_vector_type(8)))  _Float16 v8h;
typedef __attribute__((ext_vector_type(8)))  float    v8f;

namespace {
constexpr int Bn = 8, Cn = 3, Hn = 512, Wn = 512, On = 64, Kn = 9;
constexpr int CK   = 27;   // GEMM reduction dim (c*9 + k), zero-padded to 32
constexpr int KPAD = 32;
constexpr int VSTR = 40;   // LDS row stride in halfs (16B-aligned frags, conflict-spread)
}

__device__ __forceinline__ float ldx(const float* __restrict__ xc, int y, int xx) {
    return (y >= 0 && y < Hn && xx >= 0 && xx < Wn) ? xc[y * Wn + xx] : 0.0f;
}

__global__ __launch_bounds__(256)
void deform_fused(const float* __restrict__ x,
                  const float* __restrict__ w_simple,
                  const float* __restrict__ w_off,
                  const float* __restrict__ b_off,
                  const float* __restrict__ w_mask,
                  const float* __restrict__ b_mask,
                  float* __restrict__ out) {
    __shared__ __align__(16) float    sWoff[18 * 27];
    __shared__               float    sBoff[18];
    __shared__ __align__(16) float    sWmask[9 * 27];
    __shared__               float    sBmask[9];
    __shared__ __align__(16) _Float16 sWs[64][VSTR];       // f16 GEMM A: [o][ck]
    __shared__ __align__(16) _Float16 sV[8][32][VSTR];     // f16 GEMM B: [wave][pix][ck]

    const int t    = threadIdx.x;
    const int wv   = t >> 5;
    const int lane = t & 31;
    const int b    = blockIdx.z;
    const int h    = blockIdx.y;
    const int colBase = blockIdx.x * 256 + wv * 32;
    const int col  = colBase + lane;

    // ---- stage small weights in LDS (w_simple converted to f16, K padded to 32) ----
    for (int i = t; i < 18 * 27; i += 256) sWoff[i]  = w_off[i];
    for (int i = t; i <  9 * 27; i += 256) sWmask[i] = w_mask[i];
    if (t < 18) sBoff[t]  = b_off[t];
    if (t <  9) sBmask[t] = b_mask[t];
    for (int i = t; i < 64 * VSTR; i += 256) {
        int o = i / VSTR, kk = i % VSTR;
        (&sWs[0][0])[i] = (kk < CK) ? (_Float16)w_simple[o * CK + kk] : (_Float16)0.0f;
    }

    const float* xb0 = x + (size_t)b * Cn * Hn * Wn;
    // warm caches for this row (gfx1250 global_prefetch path)
    #pragma unroll
    for (int c = 0; c < Cn; ++c)
        __builtin_prefetch(xb0 + (size_t)c * Hn * Wn + (size_t)h * Wn + col, 0, 3);

    // ---- 3x3x3 neighborhood of x into registers (zero-padded borders) ----
    float xin[27];
    #pragma unroll
    for (int c = 0; c < 3; ++c)
        #pragma unroll
        for (int dy = 0; dy < 3; ++dy)
            #pragma unroll
            for (int dx = 0; dx < 3; ++dx)
                xin[c * 9 + dy * 3 + dx] = ldx(xb0 + c * Hn * Wn, h - 1 + dy, col - 1 + dx);

    __syncthreads();   // weights staged

    // zero-pad the K dimension of this pixel's B column
    #pragma unroll
    for (int kk = CK; kk < KPAD; ++kk) sV[wv][lane][kk] = (_Float16)0.0f;

    // ---- per-tap: offset/mask conv, sigmoid, bilinear sample, write f16 B column ----
    #pragma unroll
    for (int k = 0; k < Kn; ++k) {
        float wdy[27], wdx[27], wm[27];
        #pragma unroll
        for (int j = 0; j < 27; ++j) {          // contiguous rows -> ds_load_b128 packing
            wdy[j] = sWoff[(2 * k)     * 27 + j];
            wdx[j] = sWoff[(2 * k + 1) * 27 + j];
            wm[j]  = sWmask[k * 27 + j];
        }
        float ady = sBoff[2 * k], adx = sBoff[2 * k + 1], am = sBmask[k];
        #pragma unroll
        for (int j = 0; j < 27; ++j) {
            float xv = xin[j];
            ady = fmaf(wdy[j], xv, ady);
            adx = fmaf(wdx[j], xv, adx);
            am  = fmaf(wm[j],  xv, am);
        }
        const float m  = 1.0f / (1.0f + __expf(-am));
        const float py = (float)h   - 1.0f + (float)(k / 3) + ady;
        const float px = (float)col - 1.0f + (float)(k % 3) + adx;
        const float y0f = floorf(py), x0f = floorf(px);
        const float ty = py - y0f, tx = px - x0f;
        const int   y0 = (int)y0f,  x0 = (int)x0f;
        const float w00 = (1.0f - ty) * (1.0f - tx), w01 = (1.0f - ty) * tx;
        const float w10 = ty * (1.0f - tx),          w11 = ty * tx;
        #pragma unroll
        for (int c = 0; c < 3; ++c) {
            const float* xc = xb0 + c * Hn * Wn;
            float s = w00 * ldx(xc, y0,     x0)
                    + w01 * ldx(xc, y0,     x0 + 1)
                    + w10 * ldx(xc, y0 + 1, x0)
                    + w11 * ldx(xc, y0 + 1, x0 + 1);
            sV[wv][lane][c * 9 + k] = (_Float16)(m * s);
        }
    }

    __syncthreads();   // make sV visible for WMMA fragment reads

    // ---- WMMA einsum: D[16o x 16px] = A[16o x 32ck] * B[32ck x 16px] ----
    const int g = lane >> 4;     // K-half selector
    const int n = lane & 15;     // N (pixel) / M (o-row) within fragment

    v16h bf[2];
    #pragma unroll
    for (int tN = 0; tN < 2; ++tN) {
        const _Float16* p = &sV[wv][tN * 16 + n][g * 16];   // K = g*16 + e
        v8h b0 = *(const v8h*)p;
        v8h b1 = *(const v8h*)(p + 8);
        bf[tN] = __builtin_shufflevector(b0, b1, 0, 1, 2, 3, 4, 5, 6, 7,
                                                 8, 9, 10, 11, 12, 13, 14, 15);
    }

    const size_t plane = (size_t)Hn * Wn;
    #pragma unroll
    for (int oT = 0; oT < 4; ++oT) {
        // A layout: K = g*16 + e
        const _Float16* ap = &sWs[oT * 16 + n][0];
        v8h a0 = *(const v8h*)(ap + g * 16);
        v8h a1 = *(const v8h*)(ap + g * 16 + 8);
        v16h af = __builtin_shufflevector(a0, a1, 0, 1, 2, 3, 4, 5, 6, 7,
                                                  8, 9, 10, 11, 12, 13, 14, 15);
        #pragma unroll
        for (int tN = 0; tN < 2; ++tN) {
            v8f acc = {};
            acc = __builtin_amdgcn_wmma_f32_16x16x32_f16(
                      false, af, false, bf[tN], (short)0, acc, false, false);
            // D: o = oT*16 + g*8 + r ; column = colBase + tN*16 + n
            float* op = out + ((size_t)b * On + oT * 16 + g * 8) * plane
                            + (size_t)h * Wn + colBase + tN * 16 + n;
            #pragma unroll
            for (int r = 0; r < 8; ++r) op[(size_t)r * plane] = acc[r];
        }
    }
}

extern "C" void kernel_launch(void* const* d_in, const int* in_sizes, int n_in,
                              void* d_out, int out_size, void* d_ws, size_t ws_size,
                              hipStream_t stream) {
    const float* x        = (const float*)d_in[0];
    const float* w_simple = (const float*)d_in[1];
    const float* w_off    = (const float*)d_in[2];
    const float* b_off    = (const float*)d_in[3];
    const float* w_mask   = (const float*)d_in[4];
    const float* b_mask   = (const float*)d_in[5];
    float* out = (float*)d_out;

    dim3 grid(Wn / 256, Hn, Bn);   // 2 x 512 x 8 blocks, 256 threads (8 waves) each
    deform_fused<<<grid, 256, 0, stream>>>(x, w_simple, w_off, b_off, w_mask, b_mask, out);
}
